// ALIGNNSimple_2156073582917
// MI455X (gfx1250) — matrix operand order repaired and checked by reading, not compiled, not run on hardware
//
#include <hip/hip_runtime.h>
#include <math.h>

// ---------------------------------------------------------------------------
// ALIGNN forward for MI455X (gfx1250, wave32).
// Dense projections: LDS-staged GEMM on v_wmma_f32_16x16x32_bf16.
//  - A tile staged via global_load_async_to_lds_b128 (ASYNCcnt) on full tiles
//  - one staged A tile reused across the full output width (NTILES templated)
// Graph ops: coalesced float4 gathers, atomic f32 scatters, two-pass BN.
// Workspace requirement: ~440 MB.
// ---------------------------------------------------------------------------

typedef __attribute__((ext_vector_type(16))) __bf16 v16bf;
typedef __attribute__((ext_vector_type(8)))  __bf16 v8bf;
typedef __attribute__((ext_vector_type(8)))  float  v8f;
typedef __attribute__((ext_vector_type(4)))  float  v4f;

#define N_NODES   20000
#define N_EDGES   240000
#define N_LG      600000
#define N_GRAPHS  500
#define ATOM_IN   92
#define NODE_F    64
#define EDGE_F    32
#define ANGLE_F   40
#define FC_F      128
#define N_LAYERS  4
#define BN_EPS    1e-5f

__device__ __forceinline__ float softplus_f(float x) {
  if (x > 20.f)  return x;
  if (x < -20.f) return expf(x);
  return log1pf(expf(x));
}
__device__ __forceinline__ float sigmoid_f(float x) {
  return 1.f / (1.f + expf(-x));
}

// --- CDNA5 async global->LDS copy (ASYNCcnt path, CDNA5 ISA)
__device__ __forceinline__ void async_copy_b128(float* lds_dst, const float* gsrc) {
  // low 32 bits of a generic LDS address are the LDS byte offset (ISA 10.2)
  unsigned ldsoff = (unsigned)(unsigned long long)lds_dst;
  asm volatile("global_load_async_to_lds_b128 %0, %1, off"
               :: "v"(ldsoff), "v"(gsrc)
               : "memory");
}
__device__ __forceinline__ void async_wait_all() {
#if __has_builtin(__builtin_amdgcn_s_wait_asynccnt)
  __builtin_amdgcn_s_wait_asynccnt(0);
#else
  asm volatile("s_wait_asynccnt 0" ::: "memory");
#endif
}

// ------------------------------ RBF expansion ------------------------------
__global__ void rbf_edge_kernel(const float* __restrict__ r, float* __restrict__ ef, int E) {
  int e = blockIdx.x * blockDim.x + threadIdx.x;
  if (e >= E) return;
  float x = r[3 * e], y = r[3 * e + 1], z = r[3 * e + 2];
  float d = sqrtf(x * x + y * y + z * z);
  const float gamma = 31.f / 8.f;
#pragma unroll
  for (int i = 0; i < EDGE_F; ++i) {
    float c = 8.f * (float)i / 31.f;
    float t = d - c;
    ef[(long)e * EDGE_F + i] = expf(-gamma * t * t);
  }
}

__global__ void rbf_angle_kernel(const float* __restrict__ ang, float* __restrict__ af, int E) {
  int e = blockIdx.x * blockDim.x + threadIdx.x;
  if (e >= E) return;
  const float PI = 3.14159265358979323846f;
  float d = ang[e];
  const float gamma = 39.f / PI;
#pragma unroll
  for (int i = 0; i < ANGLE_F; ++i) {
    float c = -0.5f * PI + PI * (float)i / 39.f;
    float t = d - c;
    af[(long)e * ANGLE_F + i] = expf(-gamma * t * t);
  }
}

// ------------------------------ WMMA GEMM ----------------------------------
// C[M,N] = (accumulate ? C : 0) + A[M,K]*W[K,N] + bias[N], optional softplus.
// N = NTILES*16 exactly. Block: 128 threads (4 waves) -> 64 x N output tile.
// K-step 32. A tile staged fp32 in LDS (async b128 when full), B tile staged
// transposed bf16. One A fragment feeds NTILES WMMAs per K-step.
#define SA_STRIDE 36   // 64x32 fp32 tile, padded: conflict-free 16-row reads
#define SB_STRIDE 40   // Nx32 bf16 transposed tile, padded

template <int NTILES>
__global__ void wmma_gemm_kernel(const float* __restrict__ A, const float* __restrict__ W,
                                 const float* __restrict__ bias, float* __restrict__ C,
                                 int M, int K, int accumulate, int act) {
  constexpr int N = NTILES * 16;
  __shared__ float  sA[64 * SA_STRIDE];
  __shared__ __bf16 sBt[N * SB_STRIDE];

  const int tid   = threadIdx.x;
  const int lane  = tid & 31;
  const int wave  = tid >> 5;
  const int l16   = lane & 15;
  const int khalf = lane >> 4;
  const int rbase = blockIdx.y * 64;
  const int tileR = rbase + wave * 16;   // this wave's 16-row subtile

  v8f acc[NTILES];
#pragma unroll
  for (int nt = 0; nt < NTILES; ++nt)
    acc[nt] = (v8f){0.f, 0.f, 0.f, 0.f, 0.f, 0.f, 0.f, 0.f};

  if (accumulate) {
    const bool fullM = (tileR + 16 <= M);
#pragma unroll
    for (int nt = 0; nt < NTILES; ++nt) {
      int ncol = nt * 16 + l16;
#pragma unroll
      for (int v = 0; v < 8; ++v) {
        int m = tileR + v + 8 * khalf;
        acc[nt][v] = (fullM || m < M) ? C[(long)m * N + ncol] : 0.f;
      }
    }
  }

  for (int k0 = 0; k0 < K; k0 += 32) {
    // ---- stage A tile: rows rbase..rbase+63, cols k0..k0+31 (fp32) ----
    const bool fullTile = (rbase + 64 <= M) && (k0 + 32 <= K);   // block-uniform
    if (fullTile) {
#pragma unroll
      for (int i = 0; i < 4; ++i) {
        int chunk = tid + i * 128;            // 512 x float4 chunks
        int row = chunk >> 3;
        int c4  = (chunk & 7) << 2;
        async_copy_b128(&sA[row * SA_STRIDE + c4],
                        &A[(long)(rbase + row) * K + k0 + c4]);
      }
      async_wait_all();
    } else {
#pragma unroll
      for (int i = 0; i < 16; ++i) {
        int idx = tid + i * 128;              // 2048 elements
        int row = idx >> 5, col = idx & 31;
        int gr = rbase + row, gc = k0 + col;
        sA[row * SA_STRIDE + col] = (gr < M && gc < K) ? A[(long)gr * K + gc] : 0.f;
      }
    }
    // ---- stage B tile transposed + bf16: sBt[n][k] for k0..k0+31, all N ----
#pragma unroll
    for (int i = 0; i < (32 * N) / 128; ++i) {
      int idx = tid + i * 128;                // 32*N elements
      int kr = idx / N, c = idx % N;
      int gk = k0 + kr;
      float v = (gk < K) ? W[(long)gk * N + c] : 0.f;
      sBt[c * SB_STRIDE + kr] = (__bf16)v;
    }
    __syncthreads();

    // ---- A fragment: K runs [8*khalf,+8) and [16+8*khalf,+8) (ISA 7.12.2) ----
    const float* ar = &sA[(wave * 16 + l16) * SA_STRIDE];
    v4f a0 = *(const v4f*)(ar + 8 * khalf);
    v4f a1 = *(const v4f*)(ar + 8 * khalf + 4);
    v4f a2 = *(const v4f*)(ar + 16 + 8 * khalf);
    v4f a3 = *(const v4f*)(ar + 16 + 8 * khalf + 4);
    v16bf a;
#pragma unroll
    for (int i = 0; i < 4; ++i) {
      a[i]      = (__bf16)a0[i];
      a[4 + i]  = (__bf16)a1[i];
      a[8 + i]  = (__bf16)a2[i];
      a[12 + i] = (__bf16)a3[i];
    }
    // ---- one WMMA per N-subtile, reusing the A fragment ----
#pragma unroll
    for (int nt = 0; nt < NTILES; ++nt) {
      const __bf16* br = &sBt[(nt * 16 + l16) * SB_STRIDE + 8 * khalf];
      v8bf b0 = *(const v8bf*)br;
      v8bf b1 = *(const v8bf*)(br + 16);
      v16bf b;
#pragma unroll
      for (int i = 0; i < 8; ++i) { b[i] = b0[i]; b[8 + i] = b1[i]; }
      acc[nt] = __builtin_amdgcn_wmma_f32_16x16x32_bf16(
          /*neg_a=*/false, a, /*neg_b=*/false, b,
          /*c_mod=*/(short)0, acc[nt], /*reuse_a=*/false, /*reuse_b=*/false);
    }
    __syncthreads();   // protect LDS before restaging
  }

  // ---- epilogue ----
  const bool fullM = (tileR + 16 <= M);
#pragma unroll
  for (int nt = 0; nt < NTILES; ++nt) {
    int ncol = nt * 16 + l16;
    float bv = bias ? bias[ncol] : 0.f;
#pragma unroll
    for (int v = 0; v < 8; ++v) {
      int m = tileR + v + 8 * khalf;
      if (fullM || m < M) {
        float val = acc[nt][v] + bv;
        if (act == 1)      val = softplus_f(val);
        else if (act == 2) val = softplus_f(softplus_f(val));
        C[(long)m * N + ncol] = val;
      }
    }
  }
}

// ------------------------ elementwise graph kernels ------------------------
__global__ void fill_kernel(float* __restrict__ p, float v, long n) {
  long i = (long)blockIdx.x * blockDim.x + threadIdx.x;
  if (i < n) p[i] = v;
}

// m[e, f4] = PS[src[e], f4] + PD[dst[e], f4]   (float4 vectorized, F % 4 == 0)
__global__ void gather_sum4_kernel(const float* __restrict__ PS, const float* __restrict__ PD,
                                   const int* __restrict__ srcI, const int* __restrict__ dstI,
                                   float* __restrict__ Mo, int E, int F4) {
  long idx = (long)blockIdx.x * blockDim.x + threadIdx.x;
  if (idx >= (long)E * F4) return;
  int e = (int)(idx / F4);
  int f = (int)(idx % F4);
  const v4f* ps = (const v4f*)PS;
  const v4f* pd = (const v4f*)PD;
  v4f a = ps[(long)srcI[e] * F4 + f];
  v4f b = pd[(long)dstI[e] * F4 + f];
  ((v4f*)Mo)[idx] = a + b;
}

// per-feature mean / rstd over the batch axis (training-mode BatchNorm1d)
__global__ void col_stats_kernel(const float* __restrict__ X, int rows, int F,
                                 float* __restrict__ meanp, float* __restrict__ rstdp) {
  __shared__ float sh_s[256];
  __shared__ float sh_q[256];
  const int f = blockIdx.x;
  float s = 0.f, q = 0.f;
  for (int r = threadIdx.x; r < rows; r += 256) {
    float v = X[(long)r * F + f];
    s += v; q += v * v;
  }
  sh_s[threadIdx.x] = s; sh_q[threadIdx.x] = q;
  __syncthreads();
  for (int o = 128; o > 0; o >>= 1) {
    if ((int)threadIdx.x < o) {
      sh_s[threadIdx.x] += sh_s[threadIdx.x + o];
      sh_q[threadIdx.x] += sh_q[threadIdx.x + o];
    }
    __syncthreads();
  }
  if (threadIdx.x == 0) {
    float mu  = sh_s[0] / (float)rows;
    float var = sh_q[0] / (float)rows - mu * mu;
    meanp[f] = mu;
    rstdp[f] = rsqrtf(var + BN_EPS);
  }
}

// BN(m) -> split -> sigmoid(h_f)*softplus(h_s) -> atomic segment_sum into H[dst]
__global__ void gate_scatter_kernel(const float* __restrict__ Mi, const int* __restrict__ dstI,
                                    const float* __restrict__ mean, const float* __restrict__ rstd,
                                    const float* __restrict__ gm, const float* __restrict__ bm,
                                    float* __restrict__ H, int E, int nf) {
  long idx = (long)blockIdx.x * blockDim.x + threadIdx.x;
  if (idx >= (long)E * nf) return;
  int e = (int)(idx / nf);
  int f = (int)(idx % nf);
  long base = (long)e * (2 * nf);
  float hf = Mi[base + f];
  float hs = Mi[base + nf + f];
  hf = gm[f]      * (hf - mean[f])      * rstd[f]      + bm[f];
  hs = gm[nf + f] * (hs - mean[nf + f]) * rstd[nf + f] + bm[nf + f];
  float v = sigmoid_f(hf) * softplus_f(hs);
  atomicAdd(&H[(long)dstI[e] * nf + f], v);
}

// x = softplus(x + BN(h))
__global__ void node_update_kernel(float* __restrict__ X, const float* __restrict__ H,
                                   const float* __restrict__ mean, const float* __restrict__ rstd,
                                   const float* __restrict__ gn, const float* __restrict__ bnb,
                                   int rows, int nf) {
  long idx = (long)blockIdx.x * blockDim.x + threadIdx.x;
  if (idx >= (long)rows * nf) return;
  int f = (int)(idx % nf);
  float h = gn[f] * (H[idx] - mean[f]) * rstd[f] + bnb[f];
  X[idx] = softplus_f(X[idx] + h);
}

// ------------------------------- readout -----------------------------------
__global__ void pool_kernel(const float* __restrict__ X, const int* __restrict__ gid,
                            float* __restrict__ psum, float* __restrict__ pcnt, int Nn, int F) {
  long idx = (long)blockIdx.x * blockDim.x + threadIdx.x;
  if (idx >= (long)Nn * F) return;
  int i = (int)(idx / F);
  int f = (int)(idx % F);
  int g = gid[i];
  atomicAdd(&psum[(long)g * F + f], X[idx]);
  if (f == 0) atomicAdd(&pcnt[g], 1.f);
}

__global__ void pool_finalize_kernel(const float* __restrict__ psum, const float* __restrict__ pcnt,
                                     float* __restrict__ f0, int G, int F) {
  long idx = (long)blockIdx.x * blockDim.x + threadIdx.x;
  if (idx >= (long)G * F) return;
  int g = (int)(idx / F);
  float c = pcnt[g];
  if (c < 1.f) c = 1.f;
  f0[idx] = softplus_f(psum[idx] / c);
}

// out[g] = dot(feats[g,:], W_out) + b_out  (one wave32 per graph)
__global__ void out_kernel(const float* __restrict__ FE, const float* __restrict__ Wout,
                           const float* __restrict__ bout, float* __restrict__ out, int G, int F) {
  int g = blockIdx.x;
  if (g >= G) return;
  int lane = threadIdx.x;
  float s = 0.f;
  for (int f = lane; f < F; f += 32) s += FE[(long)g * F + f] * Wout[f];
#pragma unroll
  for (int o = 16; o > 0; o >>= 1) s += __shfl_xor(s, o, 32);
  if (lane == 0) out[g] = s + bout[0];
}

// ---------------------------------------------------------------------------
extern "C" void kernel_launch(void* const* d_in, const int* in_sizes, int n_in,
                              void* d_out, int out_size, void* d_ws, size_t ws_size,
                              hipStream_t stream) {
  // --- inputs (top-level dict order; params pytree in JAX sorted-key order) ---
  const float* atom = (const float*)d_in[0];   // [20000, 92]
  const float* r    = (const float*)d_in[1];   // [240000, 3]
  const float* ang  = (const float*)d_in[2];   // [600000]
  const float* Wemb = (const float*)d_in[3];   // [92, 64]
  const float* Wfc  = (const float*)d_in[4];   // [64, 128]
  const float* Wout = (const float*)d_in[5];   // [128, 1]
  const float* bemb = (const float*)d_in[6];   // [64]
  const float* bfc  = (const float*)d_in[7];   // [128]
  const float* bout = (const float*)d_in[8];   // [1]
  const float* Wd1 = (const float*)d_in[9];    // [4, 64, 128]
  const float* We1 = (const float*)d_in[10];   // [4, 32, 128]
  const float* Ws1 = (const float*)d_in[11];   // [4, 64, 128]
  const float* bd1 = (const float*)d_in[12];   // [4, 128]
  const float* be1 = (const float*)d_in[13];   // [4, 128]
  const float* bm1 = (const float*)d_in[14];   // [4, 128]
  const float* bn1 = (const float*)d_in[15];   // [4, 64]
  const float* bs1 = (const float*)d_in[16];   // [4, 128]
  const float* gm1 = (const float*)d_in[17];   // [4, 128]
  const float* gn1 = (const float*)d_in[18];   // [4, 64]
  const float* Wd2 = (const float*)d_in[19];   // [4, 32, 64]
  const float* We2 = (const float*)d_in[20];   // [4, 40, 64]
  const float* Ws2 = (const float*)d_in[21];   // [4, 32, 64]
  const float* bd2 = (const float*)d_in[22];   // [4, 64]
  const float* be2 = (const float*)d_in[23];   // [4, 64]
  const float* bm2 = (const float*)d_in[24];   // [4, 64]
  const float* bn2 = (const float*)d_in[25];   // [4, 32]
  const float* bs2 = (const float*)d_in[26];   // [4, 64]
  const float* gm2 = (const float*)d_in[27];   // [4, 64]
  const float* gn2 = (const float*)d_in[28];   // [4, 32]
  const int* src    = (const int*)d_in[29];    // [240000]
  const int* dst    = (const int*)d_in[30];    // [240000]
  const int* lg_src = (const int*)d_in[31];    // [600000]
  const int* lg_dst = (const int*)d_in[32];    // [600000]
  const int* gid    = (const int*)d_in[33];    // [20000]
  float* out = (float*)d_out;                  // [500]

  // --- workspace layout (floats), ~440 MB total ---
  float* ws = (float*)d_ws;
  float* X    = ws;                         // 1,280,000
  float* EF   = X    + 1280000;             // 7,680,000
  float* AF   = EF   + 7680000;             // 24,000,000
  float* PS   = AF   + 24000000;            // 15,360,000
  float* PD   = PS   + 15360000;            // 15,360,000
  float* MBUF = PD   + 15360000;            // 38,400,000
  float* HBUF = MBUF + 38400000;            // 7,680,000
  float* STAT = HBUF + 7680000;             // 256
  float* POOL = STAT + 256;                 // 32,000
  float* CNT  = POOL + 32000;               // 500
  float* F0   = CNT  + 500;                 // 32,000
  float* F1   = F0   + 32000;               // 64,000
  float* MEAN = STAT;
  float* RSTD = STAT + 128;

  auto gemm = [&](const float* A, const float* W, const float* b, float* C,
                  int M, int K, int N, int acc, int act) {
    dim3 grid(1, (M + 63) / 64);
    if (N == 128)
      wmma_gemm_kernel<8><<<grid, 128, 0, stream>>>(A, W, b, C, M, K, acc, act);
    else  // N == 64
      wmma_gemm_kernel<4><<<grid, 128, 0, stream>>>(A, W, b, C, M, K, acc, act);
  };
  auto ew_grid = [](long n) { return (unsigned)((n + 255) / 256); };

  // 1) RBF feature expansions
  rbf_edge_kernel<<<ew_grid(N_EDGES), 256, 0, stream>>>(r, EF, N_EDGES);
  rbf_angle_kernel<<<ew_grid(N_LG), 256, 0, stream>>>(ang, AF, N_LG);

  // 2) atom embedding
  gemm(atom, Wemb, bemb, X, N_NODES, ATOM_IN, NODE_F, 0, 0);

  // 3) message-passing layers
  for (int l = 0; l < N_LAYERS; ++l) {
    // ---- conv1: node conv over (src,dst), nf=64, F2=128 ----
    gemm(X, Ws1 + (long)l * 64 * 128, bs1 + l * 128, PS, N_NODES, 64, 128, 0, 0);
    gemm(X, Wd1 + (long)l * 64 * 128, bd1 + l * 128, PD, N_NODES, 64, 128, 0, 0);
    gather_sum4_kernel<<<ew_grid((long)N_EDGES * 32), 256, 0, stream>>>(
        PS, PD, src, dst, MBUF, N_EDGES, 32);
    gemm(EF, We1 + (long)l * 32 * 128, be1 + l * 128, MBUF, N_EDGES, 32, 128, 1, 0);
    col_stats_kernel<<<128, 256, 0, stream>>>(MBUF, N_EDGES, 128, MEAN, RSTD);
    fill_kernel<<<ew_grid((long)N_NODES * 64), 256, 0, stream>>>(HBUF, 0.f, (long)N_NODES * 64);
    gate_scatter_kernel<<<ew_grid((long)N_EDGES * 64), 256, 0, stream>>>(
        MBUF, dst, MEAN, RSTD, gm1 + l * 128, bm1 + l * 128, HBUF, N_EDGES, 64);
    col_stats_kernel<<<64, 256, 0, stream>>>(HBUF, N_NODES, 64, MEAN, RSTD);
    node_update_kernel<<<ew_grid((long)N_NODES * 64), 256, 0, stream>>>(
        X, HBUF, MEAN, RSTD, gn1 + l * 64, bn1 + l * 64, N_NODES, 64);

    // ---- conv2: line-graph conv over (lg_src,lg_dst), nf=32, F2=64 ----
    gemm(EF, Ws2 + (long)l * 32 * 64, bs2 + l * 64, PS, N_EDGES, 32, 64, 0, 0);
    gemm(EF, Wd2 + (long)l * 32 * 64, bd2 + l * 64, PD, N_EDGES, 32, 64, 0, 0);
    gather_sum4_kernel<<<ew_grid((long)N_LG * 16), 256, 0, stream>>>(
        PS, PD, lg_src, lg_dst, MBUF, N_LG, 16);
    gemm(AF, We2 + (long)l * 40 * 64, be2 + l * 64, MBUF, N_LG, 40, 64, 1, 0);
    col_stats_kernel<<<64, 256, 0, stream>>>(MBUF, N_LG, 64, MEAN, RSTD);
    fill_kernel<<<ew_grid((long)N_EDGES * 32), 256, 0, stream>>>(HBUF, 0.f, (long)N_EDGES * 32);
    gate_scatter_kernel<<<ew_grid((long)N_LG * 32), 256, 0, stream>>>(
        MBUF, lg_dst, MEAN, RSTD, gm2 + l * 64, bm2 + l * 64, HBUF, N_LG, 32);
    col_stats_kernel<<<32, 256, 0, stream>>>(HBUF, N_EDGES, 32, MEAN, RSTD);
    node_update_kernel<<<ew_grid((long)N_EDGES * 32), 256, 0, stream>>>(
        EF, HBUF, MEAN, RSTD, gn2 + l * 32, bn2 + l * 32, N_EDGES, 32);
  }

  // 4) readout
  fill_kernel<<<ew_grid(32500), 256, 0, stream>>>(POOL, 0.f, 32500);  // POOL + CNT contiguous
  pool_kernel<<<ew_grid((long)N_NODES * 64), 256, 0, stream>>>(X, gid, POOL, CNT, N_NODES, 64);
  pool_finalize_kernel<<<ew_grid((long)N_GRAPHS * 64), 256, 0, stream>>>(POOL, CNT, F0, N_GRAPHS, 64);
  gemm(F0, Wfc, bfc, F1, N_GRAPHS, 64, FC_F, 0, 2);   // softplus(softplus(fc))
  out_kernel<<<N_GRAPHS, 32, 0, stream>>>(F1, Wout, bout, out, N_GRAPHS, FC_F);
}
